// RankingSet_19911468384288
// MI455X (gfx1250) — matrix operand order repaired and checked by reading, not compiled, not run on hardware
//
#include <hip/hip_runtime.h>
#include <hip/hip_bf16.h>

// ---------------------------------------------------------------------------
// RankingSet: sims = data(N,D) @ qn^T(D,Q); ct[q] = popcount(sims[:,q] >= thr[q]
//             || isclose) - 1.   N=131072, D=2048, Q=64 (fp32 in, int32 out).
//
// MI455X plan: ~32 FLOP/byte -> memory-bound at 23.3 TB/s only if the math
// runs on the bf16 WMMA path. Two-term bf16 split (x = hi + lo, keep
// hi*hi + hi*lo + lo*hi, f32 accumulate) -> ~fp32 accuracy at bf16 matrix
// rates. Data split in-register (read exactly once from HBM); normalized
// queries pre-split to bf16 hi/lo in L2-resident workspace. One wave owns a
// 32x64 output tile: 24 v_wmma_f32_16x16x32_bf16 per K=32 chunk.
// ---------------------------------------------------------------------------

typedef __attribute__((ext_vector_type(16))) __bf16    v16bf;
typedef __attribute__((ext_vector_type(8)))  __bf16    v8bf;
typedef __attribute__((ext_vector_type(8)))  float     v8f;
typedef __attribute__((ext_vector_type(8)))  unsigned  v8u;

static __device__ __forceinline__ unsigned short f2bf_rn_bits(float f) {
    unsigned u = __builtin_bit_cast(unsigned, f);
    u += 0x7FFFu + ((u >> 16) & 1u);          // round to nearest even
    return (unsigned short)(u >> 16);
}
static __device__ __forceinline__ float bfbits2f(unsigned short s) {
    return __builtin_bit_cast(float, (unsigned)s << 16);
}
static __device__ __forceinline__ v16bf load_bf16x16(const unsigned short* p) {
    v8bf lo = *(const v8bf*)(p);
    v8bf hi = *(const v8bf*)(p + 16);
    return __builtin_shufflevector(lo, hi, 0,1,2,3,4,5,6,7,8,9,10,11,12,13,14,15);
}

// Truncation-based bf16 split of a 16-element A-fragment slice.
// Element e is K = (e<8 ? kh+e : 16+kh+e-8). Lowers to v_perm_b32 packing +
// v_fma_mix_f32_bf16 residuals (~2 VALU/element).
struct AFrag { v16bf hi, lo; };
static __device__ __forceinline__ AFrag split_afrag(const float* p, int kh) {
    uint4 r0 = *(const uint4*)(p + kh);
    uint4 r1 = *(const uint4*)(p + kh + 4);
    uint4 r2 = *(const uint4*)(p + 16 + kh);
    uint4 r3 = *(const uint4*)(p + 16 + kh + 4);
    unsigned u[16] = { r0.x, r0.y, r0.z, r0.w,  r1.x, r1.y, r1.z, r1.w,
                       r2.x, r2.y, r2.z, r2.w,  r3.x, r3.y, r3.z, r3.w };
    v8u hi, lo;
    #pragma unroll
    for (int i = 0; i < 8; ++i) {
        unsigned x0 = u[2*i], x1 = u[2*i+1];
        hi[i] = __builtin_amdgcn_perm(x1, x0, 0x07060302u);
        float l0 = __builtin_bit_cast(float, x0) -
                   __builtin_bit_cast(float, x0 & 0xFFFF0000u);
        float l1 = __builtin_bit_cast(float, x1) -
                   __builtin_bit_cast(float, x1 & 0xFFFF0000u);
        lo[i] = __builtin_amdgcn_perm(__builtin_bit_cast(unsigned, l1),
                                      __builtin_bit_cast(unsigned, l0),
                                      0x07060302u);
    }
    AFrag f;
    f.hi = __builtin_bit_cast(v16bf, hi);
    f.lo = __builtin_bit_cast(v16bf, lo);
    return f;
}

// ---------------------------------------------------------------------------
// Kernel 1: per-query preprocessing. Block q: norms, threshold, bf16 hi/lo
// RNE split of normalized query row; seeds out[q] = -1 (the "minus self"
// term + deterministic re-init every launch).
// ---------------------------------------------------------------------------
__global__ __launch_bounds__(256)
void rank_prep(const float* __restrict__ queries, const float* __restrict__ truths,
               unsigned short* __restrict__ qhi, unsigned short* __restrict__ qlo,
               float* __restrict__ thresh, int* __restrict__ out, int D) {
    const int q = blockIdx.x;
    const int t = threadIdx.x;
    __shared__ float s_q[256], s_t[256], s_d[256];
    __shared__ float s_rinv;

    const float* qr = queries + (long)q * D;
    const float* tr = truths  + (long)q * D;
    float sq = 0.f, st = 0.f, dp = 0.f;
    for (int d = t; d < D; d += 256) {
        float a = qr[d], b = tr[d];
        sq = fmaf(a, a, sq); st = fmaf(b, b, st); dp = fmaf(a, b, dp);
    }
    s_q[t] = sq; s_t[t] = st; s_d[t] = dp;
    __syncthreads();
    for (int off = 128; off > 0; off >>= 1) {
        if (t < off) { s_q[t] += s_q[t+off]; s_t[t] += s_t[t+off]; s_d[t] += s_d[t+off]; }
        __syncthreads();
    }
    if (t == 0) {
        float nq = fmaxf(sqrtf(s_q[0]), 1e-12f);
        float nt = fmaxf(sqrtf(s_t[0]), 1e-12f);
        s_rinv = 1.0f / nq;
        thresh[q] = s_d[0] / (nq * nt);
        out[q] = -1;
    }
    __syncthreads();
    const float rinv = s_rinv;
    for (int d = t; d < D; d += 256) {
        float x = qr[d] * rinv;
        unsigned short h = f2bf_rn_bits(x);
        unsigned short l = f2bf_rn_bits(x - bfbits2f(h));
        qhi[(long)q * D + d] = h;
        qlo[(long)q * D + d] = l;
    }
}

// ---------------------------------------------------------------------------
// Kernel 2: split-GEMM + threshold count. One wave = 32 data rows x all 64
// query columns. K-loop in 32-chunks, 24 bf16 WMMAs per chunk.
// Split-product order per column tile: hi*b_hi, lo*b_hi (reuses waited b_hi,
// a_lo is long since ready), then hi*b_lo (b_lo load gets maximum slack).
// ---------------------------------------------------------------------------
__global__ __launch_bounds__(256)
void rank_gemm(const float* __restrict__ data,
               const unsigned short* __restrict__ qhi,
               const unsigned short* __restrict__ qlo,
               const float* __restrict__ thresh,
               int* __restrict__ out, int N, int D) {
    __shared__ int s_cnt[64];
    if (threadIdx.x < 64) s_cnt[threadIdx.x] = 0;
    __syncthreads();

    const int wave = threadIdx.x >> 5;
    const int lane = threadIdx.x & 31;
    const long m0 = ((long)blockIdx.x * 8 + wave) * 32;
    const bool active = (m0 + 32) <= (long)N;   // wave-uniform: EXEC stays full

    v8f acc[2][4] = {{v8f{}, v8f{}, v8f{}, v8f{}},
                     {v8f{}, v8f{}, v8f{}, v8f{}}};

    if (active) {
        // A layout (16-bit 16x32): lanes 0-15 M=lane, K {kh..kh+7, 16+kh..};
        // kh = 0 (lanes 0-15) or 8 (lanes 16-31).
        const int row = lane & 15;
        const int kh  = (lane >> 4) << 3;
        // B layout (16-bit 32x16): lanes 0-15 K {0..7, 16..23}, lanes 16-31 K {8..15, 24..31}.
        const int kb  = (lane >> 4) << 3;
        const float* arow0 = data + (m0 + row)      * (long)D;
        const float* arow1 = data + (m0 + 16 + row) * (long)D;

        for (int kc = 0; kc < D; kc += 32) {
            __builtin_prefetch(arow0 + kc + 256, 0, 0);   // global_prefetch_b8
            __builtin_prefetch(arow1 + kc + 256, 0, 0);

            AFrag a0 = split_afrag(arow0 + kc, kh);
            AFrag a1 = split_afrag(arow1 + kc, kh);

            #pragma unroll
            for (int ct = 0; ct < 4; ++ct) {
                const long boff = (long)(ct * 16 + row) * D + kc + kb;
                v16bf b_hi = load_bf16x16(qhi + boff);
                v16bf b_lo = load_bf16x16(qlo + boff);
                acc[0][ct] = __builtin_amdgcn_wmma_f32_16x16x32_bf16(
                    false, a0.hi, false, b_hi, (short)0, acc[0][ct], false, false);
                acc[1][ct] = __builtin_amdgcn_wmma_f32_16x16x32_bf16(
                    false, a1.hi, false, b_hi, (short)0, acc[1][ct], false, false);
                acc[0][ct] = __builtin_amdgcn_wmma_f32_16x16x32_bf16(
                    false, a0.lo, false, b_hi, (short)0, acc[0][ct], false, false);
                acc[1][ct] = __builtin_amdgcn_wmma_f32_16x16x32_bf16(
                    false, a1.lo, false, b_hi, (short)0, acc[1][ct], false, false);
                acc[0][ct] = __builtin_amdgcn_wmma_f32_16x16x32_bf16(
                    false, a0.hi, false, b_lo, (short)0, acc[0][ct], false, false);
                acc[1][ct] = __builtin_amdgcn_wmma_f32_16x16x32_bf16(
                    false, a1.hi, false, b_lo, (short)0, acc[1][ct], false, false);
            }
        }

        // C layout: VGPR r -> row (lane<16 ? r : 8+r), col = lane&15.
        #pragma unroll
        for (int ct = 0; ct < 4; ++ct) {
            const int q = ct * 16 + (lane & 15);
            const float th  = thresh[q];
            const float tol = 1e-8f + 1e-5f * fabsf(th);  // jnp.isclose defaults
            int c = 0;
            #pragma unroll
            for (int r = 0; r < 8; ++r) {
                float s0 = acc[0][ct][r];
                float s1 = acc[1][ct][r];
                c += (s0 >= th) || (fabsf(s0 - th) <= tol);
                c += (s1 >= th) || (fabsf(s1 - th) <= tol);
            }
            atomicAdd(&s_cnt[q], c);
        }
    }
    __syncthreads();
    if (threadIdx.x < 64) {
        int v = s_cnt[threadIdx.x];
        if (v) atomicAdd(out + threadIdx.x, v);
    }
}

// ---------------------------------------------------------------------------
extern "C" void kernel_launch(void* const* d_in, const int* in_sizes, int n_in,
                              void* d_out, int out_size, void* d_ws, size_t ws_size,
                              hipStream_t stream) {
    const float* data    = (const float*)d_in[0];
    const float* queries = (const float*)d_in[1];
    const float* truths  = (const float*)d_in[2];
    int* out = (int*)d_out;

    const int Q = out_size;                 // 64
    const int D = in_sizes[1] / Q;          // 2048
    const int N = in_sizes[0] / D;          // 131072

    // Workspace: [thresh: Q floats (pad to 256B)] [qhi: Q*D u16] [qlo: Q*D u16]
    float*          thresh = (float*)d_ws;
    unsigned short* qhi    = (unsigned short*)((char*)d_ws + 256);
    unsigned short* qlo    = qhi + (size_t)Q * D;

    rank_prep<<<Q, 256, 0, stream>>>(queries, truths, qhi, qlo, thresh, out, D);

    const int rowTiles = (N + 31) / 32;     // 32 rows per wave
    const int blocks   = (rowTiles + 7) / 8;
    rank_gemm<<<blocks, 256, 0, stream>>>(data, qhi, qlo, thresh, out, N, D);
}
